// CrossTransformerBlock_5961414607457
// MI455X (gfx1250) — matrix operand order repaired and checked by reading, not compiled, not run on hardware
//
#include <hip/hip_runtime.h>

// ---------------------------------------------------------------------------
// CrossTransformerBlock for MI455X (gfx1250), compile-only target.
// All GEMM-shaped work (1x1 convs, QK^T, A*V) runs on v_wmma_f32_16x16x32_f16.
// Activations are pixel-major (B,HW,C) so both WMMA fragments load as aligned
// 16B vectors. f16 storage / f32 accumulation.
// The 1x1-conv GEMM stages its weight K-slice into LDS with
// GLOBAL_LOAD_ASYNC_TO_LDS_B128 (ASYNCcnt) so 8 waves/block share one copy.
// ---------------------------------------------------------------------------

#define HW_  16384
#define B_   4
#define DIM_ 192

typedef __attribute__((ext_vector_type(16))) _Float16 v16h;
typedef __attribute__((ext_vector_type(8)))  float    v8f;

// ---- WMMA fragment loaders (layouts per CDNA5 ISA 7.12.2, wave32) ----------
// A (16x32 f16): lanes 0-15 row M=lane, K {0..7,16..23}; lanes 16-31 same rows,
// K {8..15,24..31}. Element pairs pack into VGPRs low/high.
__device__ __forceinline__ v16h frag_a_load(const _Float16* tile, int ld) {
  const int lane = threadIdx.x & 31;
  const int row  = lane & 15;
  const int koff = (lane & 16) ? 8 : 0;
  const _Float16* p = tile + (size_t)row * ld + koff;
  uint4 lo = *reinterpret_cast<const uint4*>(p);
  uint4 hi = *reinterpret_cast<const uint4*>(p + 16);
  const _Float16* lp = reinterpret_cast<const _Float16*>(&lo);
  const _Float16* hp = reinterpret_cast<const _Float16*>(&hi);
  v16h r;
#pragma unroll
  for (int i = 0; i < 8; ++i) { r[i] = lp[i]; r[8 + i] = hp[i]; }
  return r;
}

// B (32x16 f16) where logical B[k][n] = stored[n][k] (stored row-major, ld):
// lane l: column n = l&15, holds K = (l<16 ? {0..7,16..23} : {8..15,24..31}).
// Works for both global and LDS-resident tiles (addrspace inferred).
__device__ __forceinline__ v16h frag_b_load_T(const _Float16* tile, int ld) {
  const int lane = threadIdx.x & 31;
  const int n    = lane & 15;
  const int koff = (lane & 16) ? 8 : 0;
  const _Float16* p = tile + (size_t)n * ld + koff;
  uint4 lo = *reinterpret_cast<const uint4*>(p);
  uint4 hi = *reinterpret_cast<const uint4*>(p + 16);
  const _Float16* lp = reinterpret_cast<const _Float16*>(&lo);
  const _Float16* hp = reinterpret_cast<const _Float16*>(&hi);
  v16h r;
#pragma unroll
  for (int i = 0; i < 8; ++i) { r[i] = lp[i]; r[8 + i] = hp[i]; }
  return r;
}

__device__ __forceinline__ v8f wmma_f16(v16h a, v16h b, v8f c) {
  return __builtin_amdgcn_wmma_f32_16x16x32_f16(false, a, false, b,
                                                (short)0, c, false, false);
}

// Async DMA: copy 16B from global to LDS (VFLAT encoding, ASYNCcnt-tracked).
// ldsoff = low 32 bits of generic pointer == workgroup LDS byte offset.
__device__ __forceinline__ void async_g2lds_b128(unsigned ldsoff, const void* gsrc) {
  asm volatile("global_load_async_to_lds_b128 %0, %1, off"
               :
               : "v"(ldsoff), "v"(gsrc)
               : "memory");
}
__device__ __forceinline__ void wait_asynccnt0() {
  asm volatile("s_wait_asynccnt 0" ::: "memory");
}

// ---- weight convert f32 -> f16 with zero padding to (OCp, ICp) -------------
__global__ void cvt_w_kernel(const float* __restrict__ src, _Float16* __restrict__ dst,
                             int OC, int IC, int OCp, int ICp) {
  size_t t = blockIdx.x * (size_t)blockDim.x + threadIdx.x;
  if (t >= (size_t)OCp * ICp) return;
  int ic = (int)(t % ICp);
  int oc = (int)(t / ICp);
  float v = (oc < OC && ic < IC) ? src[(size_t)oc * IC + ic] : 0.0f;
  dst[t] = (_Float16)v;
}

// ---- channel LayerNorm -> pixel-major f16 ----------------------------------
__global__ void ln_px_kernel(const float* __restrict__ x, const float* __restrict__ w,
                             const float* __restrict__ bia, _Float16* __restrict__ out,
                             int C, long cstride, long pstride) {
  size_t pg = blockIdx.x * (size_t)blockDim.x + threadIdx.x;  // B*HW pixels
  if (pg >= (size_t)B_ * HW_) return;
  int p = (int)(pg % HW_);
  int b = (int)(pg / HW_);
  const float* xb = x + (size_t)b * C * HW_ + (size_t)p * pstride;
  float s = 0.f, s2 = 0.f;
  for (int c = 0; c < C; ++c) {
    float v = xb[(size_t)c * cstride];
    s += v; s2 += v * v;
  }
  float mu = s / C;
  float var = s2 / C - mu * mu;
  float rstd = rsqrtf(var + 1e-5f);
  _Float16* o = out + pg * C;
  for (int c = 0; c < C; ++c) {
    float v = (xb[(size_t)c * cstride] - mu) * rstd * w[c] + bia[c];
    o[c] = (_Float16)v;
  }
}

// ---- WMMA pixel-major GEMM: out[p][oc] = sum_ic X[p][ic] * W[oc][ic] -------
// W K-slice (64 oc x 32 ic, 4KB) is async-DMA'd into LDS once per K-chunk and
// shared by all 8 waves of the block.
// RES_MODE: 0 none, 1 +res channel-major f32, 2 +res pixel-major f32
// OUT_MODE: 0 f32 pixel-major, 1 f16 pixel-major, 2 f32 channel-major
template <int RES_MODE, int OUT_MODE>
__global__ __launch_bounds__(256) void gemm_px_kernel(
    const _Float16* __restrict__ X, const _Float16* __restrict__ Wm,
    float* __restrict__ outF, _Float16* __restrict__ outH,
    const float* __restrict__ res, int ICp, int OCp) {
  __shared__ __align__(16) _Float16 Wl[64 * 32];  // 4KB staging tile
  const int lane = threadIdx.x & 31;
  const int wid  = threadIdx.x >> 5;
  const int b    = blockIdx.z;
  const int m0   = blockIdx.x * 128 + wid * 16;  // pixel tile (16 rows)
  const int ocb  = blockIdx.y * 64;              // 4 x 16 oc tiles
  const _Float16* Xb = X + ((size_t)b * HW_ + m0) * ICp;
  // this thread's 16B piece of the cooperative W-tile copy
  const int cp_oc = threadIdx.x >> 2;           // 0..63
  const int cp_el = (threadIdx.x & 3) * 8;      // 0,8,16,24
  const unsigned ldsoff =
      (unsigned)(size_t)(&Wl[cp_oc * 32 + cp_el]);
  v8f acc[4] = {};
  for (int k0 = 0; k0 < ICp; k0 += 32) {
    // async copy W[ocb..ocb+63][k0..k0+31] -> LDS (one b128 per thread)
    async_g2lds_b128(ldsoff, Wm + (size_t)(ocb + cp_oc) * ICp + k0 + cp_el);
    wait_asynccnt0();
    __syncthreads();
    if (k0 + 32 < ICp) __builtin_prefetch(Xb + k0 + 32, 0, 1);  // global_prefetch_b8
    v16h a = frag_a_load(Xb + k0, ICp);
#pragma unroll
    for (int ni = 0; ni < 4; ++ni) {
      v16h bf = frag_b_load_T(&Wl[(ni * 16) * 32], 32);  // ds_load_b128 path
      acc[ni] = wmma_f16(a, bf, acc[ni]);
    }
    __syncthreads();  // done reading before next overwrite
  }
#pragma unroll
  for (int ni = 0; ni < 4; ++ni) {
#pragma unroll
    for (int r = 0; r < 8; ++r) {
      int m = m0 + r + ((lane & 16) ? 8 : 0);
      int n = ocb + ni * 16 + (lane & 15);
      float v = acc[ni][r];
      if (RES_MODE == 1) v += res[((size_t)b * OCp + n) * HW_ + m];
      if (RES_MODE == 2) v += res[((size_t)b * HW_ + m) * OCp + n];
      if (OUT_MODE == 0) outF[((size_t)b * HW_ + m) * OCp + n] = v;
      if (OUT_MODE == 1) outH[((size_t)b * HW_ + m) * OCp + n] = (_Float16)v;
      if (OUT_MODE == 2) outF[((size_t)b * OCp + n) * HW_ + m] = v;
    }
  }
}

// ---- grouped depthwise 3x3, pixel-major, SAME padding ----------------------
__global__ void dw3x3_kernel(const _Float16* __restrict__ in, const float* __restrict__ w,
                             _Float16* __restrict__ out, int C, int IPG) {
  size_t t = blockIdx.x * (size_t)blockDim.x + threadIdx.x;  // B*HW*C
  if (t >= (size_t)B_ * HW_ * C) return;
  int c = (int)(t % C);
  size_t r = t / C;
  int p = (int)(r % HW_);
  int b = (int)(r / HW_);
  int y0 = p >> 7, x0 = p & 127;
  float acc = 0.f;
  for (int i = 0; i < IPG; ++i) {
    int cin = (IPG == 2) ? ((c & ~1) + i) : c;
    const float* wr = w + ((size_t)c * IPG + i) * 9;
    for (int dy = -1; dy <= 1; ++dy) {
      int yy = y0 + dy; if (yy < 0 || yy > 127) continue;
      for (int dx = -1; dx <= 1; ++dx) {
        int xx = x0 + dx; if (xx < 0 || xx > 127) continue;
        acc += wr[(dy + 1) * 3 + dx + 1] *
               (float)in[(((size_t)b * HW_) + (yy << 7) + xx) * C + cin];
      }
    }
  }
  out[t] = (_Float16)acc;
}

// ---- pixel-major -> channel-major transpose (f16) --------------------------
__global__ void transpose_px2cm_kernel(const _Float16* __restrict__ in, int ldin, int coff,
                                       _Float16* __restrict__ out, int Cc) {
  size_t t = blockIdx.x * (size_t)blockDim.x + threadIdx.x;  // B*Cc*HW, p fastest
  if (t >= (size_t)B_ * Cc * HW_) return;
  int p = (int)(t % HW_);
  size_t r = t / HW_;
  int c = (int)(r % Cc);
  int b = (int)(r / Cc);
  out[t] = in[((size_t)b * HW_ + p) * ldin + coff + c];
}

// ---- per-row inverse L2 norm (rows of length HW, channel-major) ------------
__global__ void rownorm_kernel(const _Float16* __restrict__ rows, float* __restrict__ inv) {
  __shared__ float red[256];
  const _Float16* r = rows + (size_t)blockIdx.x * HW_;
  float s = 0.f;
  for (int i = threadIdx.x; i < HW_; i += 256) {
    float v = (float)r[i];
    s += v * v;
  }
  red[threadIdx.x] = s;
  __syncthreads();
  for (int off = 128; off > 0; off >>= 1) {
    if ((int)threadIdx.x < off) red[threadIdx.x] += red[threadIdx.x + off];
    __syncthreads();
  }
  if (threadIdx.x == 0) inv[blockIdx.x] = 1.0f / fmaxf(sqrtf(red[0]), 1e-12f);
}

// ---- attention scores + softmax: S = Q K^T (K=16384 via WMMA), fold norms --
// grid.x = B*2*4 blocks; 9 waves each own one 16x16 tile of the 48x48 result.
__global__ __launch_bounds__(288) void attn_scores_kernel(
    const _Float16* __restrict__ qT, const _Float16* __restrict__ k1T,
    const _Float16* __restrict__ k2T, const float* __restrict__ qinv,
    const float* __restrict__ k1inv, const float* __restrict__ k2inv,
    const float* __restrict__ temp1, const float* __restrict__ temp2,
    _Float16* __restrict__ P) {
  __shared__ float Sf[48 * 52];
  const int blk = blockIdx.x;
  const int b = blk >> 3;
  const int branch = (blk >> 2) & 1;
  const int head = blk & 3;
  const _Float16* kT = branch ? k2T : k1T;
  const float* kinv = branch ? k2inv : k1inv;
  const float tmp = (branch ? temp2 : temp1)[head];
  const int lane = threadIdx.x & 31;
  const int wid = threadIdx.x >> 5;  // 0..8
  const int mi = wid / 3, ni = wid % 3;
  const _Float16* Abase =
      qT + ((size_t)(b * 384 + branch * 192 + head * 48 + mi * 16)) * HW_;
  const _Float16* Bbase = kT + ((size_t)(b * 192 + head * 48 + ni * 16)) * HW_;
  v8f acc = {};
  for (int k0 = 0; k0 < HW_; k0 += 32) {
    v16h a = frag_a_load(Abase + k0, HW_);
    v16h bf = frag_b_load_T(Bbase + k0, HW_);
    acc = wmma_f16(a, bf, acc);
  }
#pragma unroll
  for (int r = 0; r < 8; ++r) {
    int mm = mi * 16 + r + ((lane & 16) ? 8 : 0);
    int nn = ni * 16 + (lane & 15);
    Sf[mm * 52 + nn] = acc[r];
  }
  __syncthreads();
  int i = threadIdx.x;
  if (i < 48) {
    float qi = qinv[b * 384 + branch * 192 + head * 48 + i];
    float vals[48];
    float mx = -3.4e38f;
    for (int j = 0; j < 48; ++j) {
      float kv = kinv[b * 192 + head * 48 + j];
      float v = Sf[i * 52 + j] * tmp * qi * kv;
      vals[j] = v;
      mx = fmaxf(mx, v);
    }
    float s = 0.f;
    for (int j = 0; j < 48; ++j) { float e = expf(vals[j] - mx); vals[j] = e; s += e; }
    float invs = 1.0f / s;
    _Float16* Pr = P + (size_t)blk * 48 * 64 + (size_t)i * 64;
    for (int j = 0; j < 48; ++j) Pr[j] = (_Float16)(vals[j] * invs);
    for (int j = 48; j < 64; ++j) Pr[j] = (_Float16)0.0f;  // zero-pad K to 64
  }
}

// ---- out^T = V^T P^T (pixel-major result), write into concat buffer --------
__global__ __launch_bounds__(256) void attn_av_kernel(
    const _Float16* __restrict__ P, const _Float16* __restrict__ kv1px,
    const _Float16* __restrict__ kv2px, _Float16* __restrict__ concat) {
  const int lane = threadIdx.x & 31;
  const int wid = threadIdx.x >> 5;  // 0..7
  const int b = blockIdx.z;
  const int hb = blockIdx.y;
  const int branch = hb >> 2, head = hb & 3;
  const int p0 = (blockIdx.x * 8 + wid) * 16;
  const _Float16* vpx = branch ? kv2px : kv1px;
  const _Float16* Abase = vpx + ((size_t)b * HW_ + p0) * 384 + 192 + head * 48;
  const _Float16* Pb = P + ((size_t)((b * 2 + branch) * 4 + head)) * 48 * 64;
  v8f acc[3] = {};
#pragma unroll
  for (int k0 = 0; k0 < 64; k0 += 32) {
    v16h a = frag_a_load(Abase + k0, 384);
#pragma unroll
    for (int ci = 0; ci < 3; ++ci) {
      v16h bf = frag_b_load_T(Pb + (size_t)ci * 16 * 64 + k0, 64);
      acc[ci] = wmma_f16(a, bf, acc[ci]);
    }
  }
#pragma unroll
  for (int ci = 0; ci < 3; ++ci) {
#pragma unroll
    for (int r = 0; r < 8; ++r) {
      int m = p0 + r + ((lane & 16) ? 8 : 0);
      int c = ci * 16 + (lane & 15);
      concat[((size_t)b * HW_ + m) * 384 + branch * 192 + head * 48 + c] =
          (_Float16)acc[ci][r];
    }
  }
}

// ---- fused FFN depthwise 3x3 + exact GELU gating ---------------------------
__global__ void ffn_dw_gelu_kernel(const _Float16* __restrict__ pin,
                                   const float* __restrict__ w,
                                   _Float16* __restrict__ out) {
  size_t t = blockIdx.x * (size_t)blockDim.x + threadIdx.x;  // B*HW*512
  if (t >= (size_t)B_ * HW_ * 512) return;
  int c = (int)(t % 512);
  size_t r = t / 512;
  int p = (int)(r % HW_);
  int b = (int)(r / HW_);
  if (c >= 510) { out[t] = (_Float16)0.0f; return; }
  int y0 = p >> 7, x0 = p & 127;
  float z1 = 0.f, z2 = 0.f;
  const float* w1 = w + (size_t)c * 9;
  const float* w2 = w + (size_t)(c + 510) * 9;
  for (int dy = -1; dy <= 1; ++dy) {
    int yy = y0 + dy; if (yy < 0 || yy > 127) continue;
    for (int dx = -1; dx <= 1; ++dx) {
      int xx = x0 + dx; if (xx < 0 || xx > 127) continue;
      size_t idx = (((size_t)b * HW_) + (yy << 7) + xx) * 1024;
      int tap = (dy + 1) * 3 + dx + 1;
      z1 += w1[tap] * (float)pin[idx + c];
      z2 += w2[tap] * (float)pin[idx + c + 510];
    }
  }
  float g = 0.5f * z1 * (1.0f + erff(z1 * 0.70710678118f));
  out[t] = (_Float16)(g * z2);
}

// ---------------------------------------------------------------------------
extern "C" void kernel_launch(void* const* d_in, const int* in_sizes, int n_in,
                              void* d_out, int out_size, void* d_ws, size_t ws_size,
                              hipStream_t stream) {
  (void)in_sizes; (void)n_in; (void)out_size; (void)ws_size;
  const float* x      = (const float*)d_in[0];
  const float* kv1    = (const float*)d_in[1];
  const float* kv2    = (const float*)d_in[2];
  const float* ln1_w  = (const float*)d_in[3];
  const float* ln1_b  = (const float*)d_in[4];
  const float* lnk1_w = (const float*)d_in[5];
  const float* lnk1_b = (const float*)d_in[6];
  const float* lnk2_w = (const float*)d_in[7];
  const float* lnk2_b = (const float*)d_in[8];
  const float* ln2_w  = (const float*)d_in[9];
  const float* ln2_b  = (const float*)d_in[10];
  const float* q_w    = (const float*)d_in[11];
  const float* kv1_w  = (const float*)d_in[12];
  const float* kv2_w  = (const float*)d_in[13];
  const float* q_dw   = (const float*)d_in[14];
  const float* kv1_dw = (const float*)d_in[15];
  const float* kv2_dw = (const float*)d_in[16];
  const float* temp1  = (const float*)d_in[17];
  const float* temp2  = (const float*)d_in[18];
  const float* proj_w = (const float*)d_in[19];
  const float* pin_w  = (const float*)d_in[20];
  const float* ffn_dw = (const float*)d_in[21];
  const float* pout_w = (const float*)d_in[22];
  float* outp = (float*)d_out;

  char* base = (char*)d_ws;
  size_t off = 0;
  auto alloc = [&](size_t bytes) -> char* {
    char* p = base + off;
    off = (off + bytes + 255) & ~(size_t)255;
    return p;
  };
  const size_t PX192 = (size_t)B_ * HW_ * 192 * 2;  // 25,165,824 B (f16)
  const size_t PX384 = (size_t)B_ * HW_ * 384 * 2;  // 50,331,648 B (f16)

  _Float16* wq    = (_Float16*)alloc(384 * 192 * 2);
  _Float16* wkv1  = (_Float16*)alloc(384 * 192 * 2);
  _Float16* wkv2  = (_Float16*)alloc(384 * 192 * 2);
  _Float16* wproj = (_Float16*)alloc(192 * 384 * 2);
  _Float16* wpin  = (_Float16*)alloc(1024 * 192 * 2);
  _Float16* wpout = (_Float16*)alloc(192 * 512 * 2);
  float* qinv  = (float*)alloc(B_ * 384 * 4);
  float* k1inv = (float*)alloc(B_ * 192 * 4);
  float* k2inv = (float*)alloc(B_ * 192 * 4);
  _Float16* Pbuf = (_Float16*)alloc(32 * 48 * 64 * 2);
  char* regionA = alloc(3 * PX192);           // xn,k1n,k2n ; later x2n + concat
  char* regionB = alloc(3 * PX384);           // q/kv pre-dw ; later pinout
  char* regionC = alloc(3 * PX384 + 4096);    // q/kv post-dw ; later y (gated)
  _Float16* qT  = (_Float16*)alloc(PX384);
  _Float16* k1T = (_Float16*)alloc(PX192);
  _Float16* k2T = (_Float16*)alloc(PX192);
  float* x2 = (float*)alloc((size_t)B_ * HW_ * 192 * 4);

  _Float16* xn  = (_Float16*)(regionA);
  _Float16* k1n = (_Float16*)(regionA + PX192);
  _Float16* k2n = (_Float16*)(regionA + 2 * PX192);
  _Float16* x2n    = (_Float16*)(regionA);             // reuse after q/kv GEMMs
  _Float16* concat = (_Float16*)(regionA + PX192);     // reuse (k1n/k2n dead)
  _Float16* qpre   = (_Float16*)(regionB);
  _Float16* kv1pre = (_Float16*)(regionB + PX384);
  _Float16* kv2pre = (_Float16*)(regionB + 2 * PX384);
  _Float16* pinout = (_Float16*)(regionB);             // reuse after depthwise
  _Float16* qpost   = (_Float16*)(regionC);
  _Float16* kv1post = (_Float16*)(regionC + PX384);
  _Float16* kv2post = (_Float16*)(regionC + 2 * PX384);
  _Float16* ybuf    = (_Float16*)(regionC);            // reuse after attention

  const int TPB = 256;
  auto blocks = [](size_t n, int tpb) { return (unsigned)((n + tpb - 1) / tpb); };

  // 1) weights -> f16 (zero padded)
  cvt_w_kernel<<<blocks(384 * 192, TPB), TPB, 0, stream>>>(q_w, wq, 384, 192, 384, 192);
  cvt_w_kernel<<<blocks(384 * 192, TPB), TPB, 0, stream>>>(kv1_w, wkv1, 384, 192, 384, 192);
  cvt_w_kernel<<<blocks(384 * 192, TPB), TPB, 0, stream>>>(kv2_w, wkv2, 384, 192, 384, 192);
  cvt_w_kernel<<<blocks(192 * 384, TPB), TPB, 0, stream>>>(proj_w, wproj, 192, 384, 192, 384);
  cvt_w_kernel<<<blocks(1024 * 192, TPB), TPB, 0, stream>>>(pin_w, wpin, 1020, 192, 1024, 192);
  cvt_w_kernel<<<blocks(192 * 512, TPB), TPB, 0, stream>>>(pout_w, wpout, 192, 510, 192, 512);

  // 2) LayerNorms (channel-major f32 in -> pixel-major f16 out)
  ln_px_kernel<<<blocks((size_t)B_ * HW_, TPB), TPB, 0, stream>>>(x, ln1_w, ln1_b, xn, 192, HW_, 1);
  ln_px_kernel<<<blocks((size_t)B_ * HW_, TPB), TPB, 0, stream>>>(kv1, lnk1_w, lnk1_b, k1n, 192, HW_, 1);
  ln_px_kernel<<<blocks((size_t)B_ * HW_, TPB), TPB, 0, stream>>>(kv2, lnk2_w, lnk2_b, k2n, 192, HW_, 1);

  // 3) 1x1 conv GEMMs (WMMA + async LDS weight staging): 192 -> 384
  dim3 g384(HW_ / 128, 384 / 64, B_);
  gemm_px_kernel<0, 1><<<g384, 256, 0, stream>>>(xn, wq, nullptr, qpre, nullptr, 192, 384);
  gemm_px_kernel<0, 1><<<g384, 256, 0, stream>>>(k1n, wkv1, nullptr, kv1pre, nullptr, 192, 384);
  gemm_px_kernel<0, 1><<<g384, 256, 0, stream>>>(k2n, wkv2, nullptr, kv2pre, nullptr, 192, 384);

  // 4) grouped depthwise 3x3
  size_t ndw = (size_t)B_ * HW_ * 384;
  dw3x3_kernel<<<blocks(ndw, TPB), TPB, 0, stream>>>(qpre, q_dw, qpost, 384, 2);
  dw3x3_kernel<<<blocks(ndw, TPB), TPB, 0, stream>>>(kv1pre, kv1_dw, kv1post, 384, 1);
  dw3x3_kernel<<<blocks(ndw, TPB), TPB, 0, stream>>>(kv2pre, kv2_dw, kv2post, 384, 1);

  // 5) channel-major copies for QK^T (q: all 384; k: first 192 of each kv)
  transpose_px2cm_kernel<<<blocks((size_t)B_ * 384 * HW_, TPB), TPB, 0, stream>>>(qpost, 384, 0, qT, 384);
  transpose_px2cm_kernel<<<blocks((size_t)B_ * 192 * HW_, TPB), TPB, 0, stream>>>(kv1post, 384, 0, k1T, 192);
  transpose_px2cm_kernel<<<blocks((size_t)B_ * 192 * HW_, TPB), TPB, 0, stream>>>(kv2post, 384, 0, k2T, 192);

  // 6) inverse L2 row norms (folded into softmax scale)
  rownorm_kernel<<<B_ * 384, 256, 0, stream>>>(qT, qinv);
  rownorm_kernel<<<B_ * 192, 256, 0, stream>>>(k1T, k1inv);
  rownorm_kernel<<<B_ * 192, 256, 0, stream>>>(k2T, k2inv);

  // 7) attention scores + softmax -> P (48x64 f16, zero-padded)
  attn_scores_kernel<<<B_ * 2 * 4, 288, 0, stream>>>(qT, k1T, k2T, qinv, k1inv,
                                                     k2inv, temp1, temp2, Pbuf);

  // 8) A*V (WMMA) -> concat pixel-major (B,HW,384)
  attn_av_kernel<<<dim3(HW_ / 128, 8, B_), 256, 0, stream>>>(Pbuf, kv1post, kv2post, concat);

  // 9) proj 1x1 (384->192) + residual(x, channel-major) -> x2 (pixel-major f32)
  gemm_px_kernel<1, 0><<<dim3(HW_ / 128, 3, B_), 256, 0, stream>>>(
      concat, wproj, x2, nullptr, x, 384, 192);

  // 10) LN2 on x2 (pixel-major f32 in)
  ln_px_kernel<<<blocks((size_t)B_ * HW_, TPB), TPB, 0, stream>>>(x2, ln2_w, ln2_b, x2n, 192, 1, 192);

  // 11) pin 1x1 (192->1024 padded)
  gemm_px_kernel<0, 1><<<dim3(HW_ / 128, 16, B_), 256, 0, stream>>>(
      x2n, wpin, nullptr, pinout, nullptr, 192, 1024);

  // 12) fused FFN depthwise + GELU gate -> y (B,HW,512 padded)
  ffn_dw_gelu_kernel<<<blocks((size_t)B_ * HW_ * 512, TPB), TPB, 0, stream>>>(pinout, ffn_dw, ybuf);

  // 13) pout 1x1 (512->192) + residual(x2) -> d_out channel-major f32
  gemm_px_kernel<2, 2><<<dim3(HW_ / 128, 3, B_), 256, 0, stream>>>(
      ybuf, wpout, outp, nullptr, x2, 512, 192);
}
